// Generator_63007170232336
// MI455X (gfx1250) — matrix operand order repaired and checked by reading, not compiled, not run on hardware
//
#include <hip/hip_runtime.h>
#include <hip/hip_bf16.h>
#include <math.h>

// ---------------------------------------------------------------------------
// Problem constants (B,H,F,T) = (1024, 1024, 64, 48)
// ---------------------------------------------------------------------------
#define B_   1024
#define H_   1024
#define F_   64
#define T_   48
#define KTOT 1088          // F_ + H_ (combined GEMM K dimension, valid part)
#define KPAD 1152          // padded to a multiple of 128 (9 chunks of 128)
#define KCH  128           // K chunk staged in LDS per barrier round
#define LDST 136           // LDS row stride in elements (272B: kills bank dup)
#define N4   4096          // 4*H_ (i,f,g,o)

typedef __bf16 bf16;
typedef __attribute__((ext_vector_type(16))) __bf16 v16bf;
typedef __attribute__((ext_vector_type(8)))  __bf16 v8bf;
typedef __attribute__((ext_vector_type(8)))  float  v8f;

union AV { v16bf v; v8bf h[2]; };

__device__ __forceinline__ float sigmoidf_(float x) {
    return 1.0f / (1.0f + __expf(-x));
}

__device__ __forceinline__ void loadB4(AV& b0, AV& b1, AV& b2, AV& b3,
                                       const bf16* __restrict__ r0,
                                       const bf16* __restrict__ r1,
                                       const bf16* __restrict__ r2,
                                       const bf16* __restrict__ r3, int k) {
    b0.h[0] = *(const v8bf*)(r0 + k);  b0.h[1] = *(const v8bf*)(r0 + k + 8);
    b1.h[0] = *(const v8bf*)(r1 + k);  b1.h[1] = *(const v8bf*)(r1 + k + 8);
    b2.h[0] = *(const v8bf*)(r2 + k);  b2.h[1] = *(const v8bf*)(r2 + k + 8);
    b3.h[0] = *(const v8bf*)(r3 + k);  b3.h[1] = *(const v8bf*)(r3 + k + 8);
}

__device__ __forceinline__ void loadA2(AV& a0, AV& a1, const bf16* lds,
                                       int arow, int abase, int kk) {
    const bf16* la0 = lds + arow * LDST + kk + abase;
    const bf16* la1 = lds + (16 + arow) * LDST + kk + abase;
    a0.h[0] = *(const v8bf*)(la0);  a0.h[1] = *(const v8bf*)(la0 + 8);
    a1.h[0] = *(const v8bf*)(la1);  a1.h[1] = *(const v8bf*)(la1 + 8);
}

__device__ __forceinline__ void burst8(v8f* acc, const AV& a0, const AV& a1,
                                       const AV& b0, const AV& b1,
                                       const AV& b2, const AV& b3) {
    acc[0] = __builtin_amdgcn_wmma_f32_16x16x32_bf16(false, a0.v, false, b0.v,
                                                     (short)0, acc[0], false, false);
    acc[1] = __builtin_amdgcn_wmma_f32_16x16x32_bf16(false, a1.v, false, b0.v,
                                                     (short)0, acc[1], false, false);
    acc[2] = __builtin_amdgcn_wmma_f32_16x16x32_bf16(false, a0.v, false, b1.v,
                                                     (short)0, acc[2], false, false);
    acc[3] = __builtin_amdgcn_wmma_f32_16x16x32_bf16(false, a1.v, false, b1.v,
                                                     (short)0, acc[3], false, false);
    acc[4] = __builtin_amdgcn_wmma_f32_16x16x32_bf16(false, a0.v, false, b2.v,
                                                     (short)0, acc[4], false, false);
    acc[5] = __builtin_amdgcn_wmma_f32_16x16x32_bf16(false, a1.v, false, b2.v,
                                                     (short)0, acc[5], false, false);
    acc[6] = __builtin_amdgcn_wmma_f32_16x16x32_bf16(false, a0.v, false, b3.v,
                                                     (short)0, acc[6], false, false);
    acc[7] = __builtin_amdgcn_wmma_f32_16x16x32_bf16(false, a1.v, false, b3.v,
                                                     (short)0, acc[7], false, false);
}

// ---------------------------------------------------------------------------
// One-time: Wt[n][k] (bf16, K padded): k<64 -> kernel[k][n],
// k<1088 -> rkernel[k-64][n], else 0.
// ---------------------------------------------------------------------------
__global__ void prep_weights(const float* __restrict__ kern,
                             const float* __restrict__ rkern,
                             bf16* __restrict__ Wt) {
    int idx = blockIdx.x * 256 + threadIdx.x;
    if (idx >= N4 * KPAD) return;
    int n = idx / KPAD;
    int k = idx - n * KPAD;
    float v = 0.0f;
    if (k < F_)        v = kern[(size_t)k * N4 + n];
    else if (k < KTOT) v = rkern[(size_t)(k - F_) * N4 + n];
    Wt[idx] = (bf16)v;
}

// One-time: zero both A buffers entirely (h0 == 0, K pad == 0).
__global__ void zero_abufs(bf16* __restrict__ A0, bf16* __restrict__ A1) {
    int idx = blockIdx.x * 256 + threadIdx.x;
    if (idx >= B_ * KPAD) return;
    A0[idx] = (bf16)0.0f;
    A1[idx] = (bf16)0.0f;
}

// One-time: state = context_state[:,2,:], c = 0.
__global__ void init_state(const float* __restrict__ ctx,
                           float* __restrict__ state,
                           float* __restrict__ cbuf) {
    int idx = blockIdx.x * 256 + threadIdx.x;
    if (idx >= B_ * H_) return;
    int b = idx / H_;
    int h = idx - b * H_;
    state[idx] = ctx[(size_t)b * 3 * H_ + 2 * H_ + h];
    cbuf[idx]  = 0.0f;
}

// ---------------------------------------------------------------------------
// Per step: mu/sigma GEMV + act; one wave32 per batch row.
// ---------------------------------------------------------------------------
__global__ void act_kernel(const float* __restrict__ state,
                           const float* __restrict__ w1, const float* __restrict__ b1,
                           const float* __restrict__ w2, const float* __restrict__ b2,
                           const float* __restrict__ eps, int t,
                           bf16* __restrict__ Ar,
                           float* __restrict__ mus, float* __restrict__ sigmas,
                           float* __restrict__ acts) {
    int wave = threadIdx.x >> 5;
    int lane = threadIdx.x & 31;
    int b = blockIdx.x * 8 + wave;

    const float* s = state + (size_t)b * H_;
    float s1 = 0.0f, s2 = 0.0f;
    #pragma unroll 4
    for (int j = lane; j < H_; j += 32) {
        float v = s[j];
        s1 = fmaf(v, w1[j], s1);
        s2 = fmaf(v, w2[j], s2);
    }
    #pragma unroll
    for (int off = 16; off > 0; off >>= 1) {
        s1 += __shfl_xor(s1, off, 32);
        s2 += __shfl_xor(s2, off, 32);
    }
    float mu = fmaxf(s1 + b1[0], 0.0f);
    float r2 = fmaxf(s2 + b2[0], 0.0f);
    float sg = ((r2 > 20.0f) ? r2 : log1pf(__expf(r2))) + 1e-8f;

    if (lane == 0) {
        mus[(size_t)t * B_ + b]    = mu;
        sigmas[(size_t)t * B_ + b] = sg;
    }

    const float* ep = eps + ((size_t)t * B_ + b) * F_;
    #pragma unroll
    for (int f = lane; f < F_; f += 32) {
        float a = mu + sg * ep[f];
        a = fminf(fmaxf(a, 0.0f), 1.0f);
        acts[((size_t)b * T_ + t) * F_ + f] = a;           // (B,T,F) layout
        Ar[(size_t)b * KPAD + f] = (bf16)a;
    }
}

// ---------------------------------------------------------------------------
// Per step: Z = A @ Wt^T + bias; gates; c/h update; h_{t+1} -> other A buffer.
// Block = 32 M-rows x 128 hidden cols, 8 waves; each wave 8 accumulators
// (i/f/g/o x 2 M-subtiles). B fragments are software-pipelined one K-step
// ahead in a rotating two-buffer (prefetch crosses LDS chunk barriers since
// B never touches LDS), so each 8-WMMA burst consumes loads issued a full
// burst earlier. A staged in LDS in 128-wide K chunks.
// ---------------------------------------------------------------------------
__global__ void __launch_bounds__(256)
gemm_gate(const bf16* __restrict__ Ar, const bf16* __restrict__ Wt,
          const float* __restrict__ bias,
          float* __restrict__ cbuf, float* __restrict__ state,
          bf16* __restrict__ Aw) {
    __shared__ bf16 ldsA[32 * LDST];                // 8.5 KB

    int wave = threadIdx.x >> 5;
    int lane = threadIdx.x & 31;
    int m0 = blockIdx.x * 32;                       // 32 M tiles
    int n  = blockIdx.y * 128 + wave * 16 + (lane & 15);  // hidden col 0..1023
    int halfK = (lane >> 4) * 16;                   // B frag K sub-half
    int arow  = lane & 15;
    int abase = (lane < 16) ? 0 : 16;               // A frag K chunk select

    v8f acc[8] = {};                                // [gate*2 + msub]

    const bf16* brow0 = Wt + (size_t)(0 * H_ + n) * KPAD + halfK;
    const bf16* brow1 = Wt + (size_t)(1 * H_ + n) * KPAD + halfK;
    const bf16* brow2 = Wt + (size_t)(2 * H_ + n) * KPAD + halfK;
    const bf16* brow3 = Wt + (size_t)(3 * H_ + n) * KPAD + halfK;

    // Staging coords: 8 threads per row, 32B (16 elems) each
    int srow = threadIdx.x >> 3;                    // 0..31
    int scol = (threadIdx.x & 7) * 16;              // 0,16,...,112

    // Prime the B pipeline with k = 0.
    AV c0, c1, c2, c3, x0, x1, x2, x3;              // rotating 2-buffer
    loadB4(c0, c1, c2, c3, brow0, brow1, brow2, brow3, 0);

    for (int kc = 0; kc < KPAD; kc += KCH) {
        const bf16* gsrc = &Ar[(size_t)(m0 + srow) * KPAD + kc + scol];
        bf16* ldst = &ldsA[srow * LDST + scol];
        *(uint4*)(ldst)     = *(const uint4*)(gsrc);
        *(uint4*)(ldst + 8) = *(const uint4*)(gsrc + 8);
        __syncthreads();

        AV a0, a1;
        // step 0: compute k=kc+0 (cur), prefetch kc+32 (nxt)
        loadA2(a0, a1, ldsA, arow, abase, 0);
        loadB4(x0, x1, x2, x3, brow0, brow1, brow2, brow3, kc + 32);
        burst8(acc, a0, a1, c0, c1, c2, c3);

        // step 1: compute kc+32 (nxt), prefetch kc+64 (cur)
        loadA2(a0, a1, ldsA, arow, abase, 32);
        loadB4(c0, c1, c2, c3, brow0, brow1, brow2, brow3, kc + 64);
        burst8(acc, a0, a1, x0, x1, x2, x3);

        // step 2: compute kc+64 (cur), prefetch kc+96 (nxt)
        loadA2(a0, a1, ldsA, arow, abase, 64);
        loadB4(x0, x1, x2, x3, brow0, brow1, brow2, brow3, kc + 96);
        burst8(acc, a0, a1, c0, c1, c2, c3);

        // step 3: compute kc+96 (nxt), prefetch next chunk's k (cur)
        int kc2 = (kc + KCH < KPAD) ? (kc + KCH) : 0;   // clamp: stay in-bounds
        loadA2(a0, a1, ldsA, arow, abase, 96);
        loadB4(c0, c1, c2, c3, brow0, brow1, brow2, brow3, kc2);
        burst8(acc, a0, a1, x0, x1, x2, x3);

        __syncthreads();
    }

    // Epilogue: gates + LSTM cell update, in-register.
    float bi = bias[n];
    float bf = bias[H_ + n];
    float bg = bias[2 * H_ + n];
    float bo = bias[3 * H_ + n];

    #pragma unroll
    for (int msub = 0; msub < 2; ++msub) {
        #pragma unroll
        for (int r = 0; r < 8; ++r) {
            int M = m0 + msub * 16 + r + ((lane < 16) ? 0 : 8);
            size_t idx = (size_t)M * H_ + n;
            float zi = acc[0 + msub][r] + bi;
            float zf = acc[2 + msub][r] + bf;
            float zg = acc[4 + msub][r] + bg;
            float zo = acc[6 + msub][r] + bo;
            float co = cbuf[idx];
            float cn = sigmoidf_(zf) * co + sigmoidf_(zi) * tanhf(zg);
            float hn = sigmoidf_(zo) * tanhf(cn);
            cbuf[idx]  = cn;
            state[idx] = hn;                            // f32 state for GEMV
            Aw[(size_t)M * KPAD + F_ + n] = (bf16)hn;   // bf16 h for next GEMM
        }
    }
}

// ---------------------------------------------------------------------------
// Host launch: prep + init once, then T sequential (act, gemm_gate) pairs.
// ---------------------------------------------------------------------------
extern "C" void kernel_launch(void* const* d_in, const int* in_sizes, int n_in,
                              void* d_out, int out_size, void* d_ws, size_t ws_size,
                              hipStream_t stream) {
    const float* ctx   = (const float*)d_in[0];
    const float* w1    = (const float*)d_in[1];
    const float* b1    = (const float*)d_in[2];
    const float* w2    = (const float*)d_in[3];
    const float* b2    = (const float*)d_in[4];
    const float* kern  = (const float*)d_in[5];
    const float* rkern = (const float*)d_in[6];
    const float* bias  = (const float*)d_in[7];
    const float* eps   = (const float*)d_in[8];

    char* ws = (char*)d_ws;
    size_t o = 0;
    bf16*  Wt = (bf16*)(ws + o);  o += (size_t)N4 * KPAD * sizeof(bf16);   // 9.4 MB
    bf16*  A0 = (bf16*)(ws + o);  o += (size_t)B_ * KPAD * sizeof(bf16);   // 2.4 MB
    bf16*  A1 = (bf16*)(ws + o);  o += (size_t)B_ * KPAD * sizeof(bf16);   // 2.4 MB
    float* st = (float*)(ws + o); o += (size_t)B_ * H_ * sizeof(float);    // 4.2 MB
    float* cb = (float*)(ws + o);                                          // 4.2 MB

    float* mus    = (float*)d_out;
    float* sigmas = mus + (size_t)T_ * B_;
    float* acts   = sigmas + (size_t)T_ * B_;

    prep_weights<<<(N4 * KPAD + 255) / 256, 256, 0, stream>>>(kern, rkern, Wt);
    zero_abufs<<<(B_ * KPAD + 255) / 256, 256, 0, stream>>>(A0, A1);
    init_state<<<(B_ * H_ + 255) / 256, 256, 0, stream>>>(ctx, st, cb);

    for (int t = 0; t < T_; ++t) {
        bf16* Ar = (t & 1) ? A1 : A0;   // holds h_t; act_t written into it now
        bf16* Aw = (t & 1) ? A0 : A1;   // receives h_{t+1}
        act_kernel<<<B_ / 8, 256, 0, stream>>>(st, w1, b1, w2, b2, eps, t,
                                               Ar, mus, sigmas, acts);
        dim3 grid(32, 8);               // 32 M-tiles(32 rows) x 8 N-blocks(128)
        gemm_gate<<<grid, 256, 0, stream>>>(Ar, Wt, bias, cb, st, Aw);
    }
}
